// DotProductAttention_88201448390668
// MI455X (gfx1250) — matrix operand order repaired and checked by reading, not compiled, not run on hardware
//
#include <hip/hip_runtime.h>

typedef __attribute__((ext_vector_type(16))) _Float16 v16h;
typedef __attribute__((ext_vector_type(8)))  float    v8f;

namespace {
constexpr int kB  = 4;     // batch
constexpr int kH  = 16;    // heads
constexpr int kS  = 2048;  // sequence length
constexpr int kD  = 64;    // head dim
constexpr int BM  = 128;   // query rows per block (8 waves x 16 rows)
constexpr int BN  = 64;    // keys per main-loop iteration
constexpr int WAVES    = BM / 16;       // 8
constexpr int NTHREADS = WAVES * 32;    // 256 (wave32)
constexpr int NKT      = kS / BN;       // 32 key tiles
constexpr int kBH      = kB * kH;       // 64
}

static __device__ __forceinline__ v8f wmma_f16(v16h a, v16h b, v8f c) {
  // D = A(16x32 f16) * B(32x16 f16) + C(16x16 f32)
  return __builtin_amdgcn_wmma_f32_16x16x32_f16(false, a, false, b,
                                                (short)0, c, false, false);
}

// ---------------------------------------------------------------------------
// Pre-pass: K -> f16 row-major copy; V -> f16 transposed [bh][d][s].
// Bandwidth-bound one-shot (67 MB read / 33.5 MB write ~ 4 us at 23.3 TB/s);
// makes main-loop tiles raw-byte-copyable for async-to-LDS.
// ---------------------------------------------------------------------------
__launch_bounds__(256, 2)
__global__ void cvt_kv(const float* __restrict__ Kg, const float* __restrict__ Vg,
                       _Float16* __restrict__ Kh, _Float16* __restrict__ Vt)
{
  __shared__ float tile[64][65];   // padded: conflict-free transpose
  const int bh = blockIdx.y;
  const int s0 = blockIdx.x * 64;
  const int tid = threadIdx.x;
  const float*    Kb  = Kg + (size_t)bh * kS * kD;
  const float*    Vb  = Vg + (size_t)bh * kS * kD;
  _Float16*       Khb = Kh + (size_t)bh * kS * kD;
  _Float16*       Vtb = Vt + (size_t)bh * kD * kS;
  for (int it = tid; it < 64 * 64; it += 256) {
    const int s = it >> 6, d = it & 63;
    Khb[(size_t)(s0 + s) * kD + d] = (_Float16)Kb[(size_t)(s0 + s) * kD + d];
    tile[s][d] = Vb[(size_t)(s0 + s) * kD + d];
  }
  __syncthreads();
  for (int it = tid; it < 64 * 64; it += 256) {
    const int d = it >> 6, s = it & 63;
    Vtb[(size_t)d * kS + s0 + s] = (_Float16)tile[s][d];
  }
}

// ---------------------------------------------------------------------------
// Main flash-attention kernel: double-buffered async global->LDS staging
// (ASYNCcnt) overlapped with f16 WMMA compute.
// ---------------------------------------------------------------------------
__launch_bounds__(NTHREADS, 1)
__global__ void fa_fwd_async(const float* __restrict__ Qg,
                             const _Float16* __restrict__ Kh,
                             const _Float16* __restrict__ Vt,
                             const unsigned char* __restrict__ Mg,
                             float* __restrict__ Og)
{
  __shared__ _Float16     Ks[2][BN][kD];        // 16 KB  [buf][key][dim]
  __shared__ _Float16     Vs[2][kD][BN];        // 16 KB  [buf][dim][key]
  __shared__ unsigned int Ms[2][BM * BN / 4];   // 16 KB  mask bytes as dwords
  __shared__ _Float16     Ps[WAVES][16][BN];    // 16 KB  per-wave P relayout

  const int tid  = threadIdx.x;
  const int wave = tid >> 5;
  const int lane = tid & 31;
  const int hgrp = lane >> 4;
  const int l16  = lane & 15;

  const int bh    = blockIdx.y;
  const int b     = bh / kH;
  const int qblk  = blockIdx.x * BM;
  const int qrow0 = qblk + wave * 16;

  const float*         Qb  = Qg + (size_t)bh * kS * kD;
  const _Float16*      Khb = Kh + (size_t)bh * kS * kD;   // f16 row-major
  const _Float16*      Vtb = Vt + (size_t)bh * kD * kS;   // f16 [d][s]
  const unsigned char* Mb  = Mg + (size_t)b * kS * kS;
  float*               Ob  = Og + (size_t)bh * kS * kD;

  // LDS byte offsets (generic addr low 32 bits == LDS offset on CDNA5)
  const unsigned int ksAddr = (unsigned int)(unsigned long long)&Ks[0][0][0];
  const unsigned int vsAddr = (unsigned int)(unsigned long long)&Vs[0][0][0];
  const unsigned int msAddr = (unsigned int)(unsigned long long)&Ms[0][0];

  // Issue one tile's staging: 6 async b128 copies per thread (ASYNCcnt += 6/wave).
  auto stage = [&](int kt2, int buf) {
    const int kbase2 = kt2 * BN;
    const unsigned int c = (unsigned int)tid;
    {
      // K tile: 8 KB contiguous (rows of 128 B, consecutive keys). Note offset:
      // the immediate adds to BOTH the LDS and global address (ISA 10.x / 8.x).
      const unsigned int lds = ksAddr + (unsigned)buf * 8192u + c * 16u;
      const unsigned int off = (unsigned)(kbase2 * kD * 2) + c * 16u;
      asm volatile("global_load_async_to_lds_b128 %0, %1, %2 offset:0"
                   :: "v"(lds), "v"(off), "s"(Khb) : "memory");
      asm volatile("global_load_async_to_lds_b128 %0, %1, %2 offset:4096"
                   :: "v"(lds), "v"(off), "s"(Khb) : "memory");
    }
    {
      // V tile: 64 rows (dims) of 128 B, row stride kS*2 bytes in Vt.
      const unsigned int lds  = vsAddr + (unsigned)buf * 8192u + c * 16u;
      const unsigned int off  = (c >> 3) * (unsigned)(kS * 2)
                              + (unsigned)(kbase2 * 2) + (c & 7u) * 16u;
      const unsigned int lds2 = lds + 4096u;
      const unsigned int off2 = off + 32u * (unsigned)(kS * 2);
      asm volatile("global_load_async_to_lds_b128 %0, %1, %2 offset:0"
                   :: "v"(lds), "v"(off), "s"(Vtb) : "memory");
      asm volatile("global_load_async_to_lds_b128 %0, %1, %2 offset:0"
                   :: "v"(lds2), "v"(off2), "s"(Vtb) : "memory");
    }
    {
      // mask tile: 128 rows of 64 B, row stride kS bytes.
      const unsigned int lds  = msAddr + (unsigned)buf * 8192u + c * 16u;
      const unsigned int off  = (unsigned)((qblk + (int)(c >> 2)) * kS)
                              + (unsigned)kbase2 + (c & 3u) * 16u;
      const unsigned int lds2 = lds + 4096u;
      const unsigned int off2 = off + 64u * (unsigned)kS;
      asm volatile("global_load_async_to_lds_b128 %0, %1, %2 offset:0"
                   :: "v"(lds), "v"(off), "s"(Mb) : "memory");
      asm volatile("global_load_async_to_lds_b128 %0, %1, %2 offset:0"
                   :: "v"(lds2), "v"(off2), "s"(Mb) : "memory");
    }
  };

  // ---- Q tile (16 rows/wave) packed once into WMMA A-fragment layout ----
  v16h qa[2];
  {
    const float* qrow = Qb + (size_t)(qrow0 + l16) * kD;
#pragma unroll
    for (int f = 0; f < 2; ++f)
#pragma unroll
      for (int j = 0; j < 8; ++j) {
        const int k = f * 32 + hgrp * 16 + j * 2;
        qa[f][2 * j]     = (_Float16)qrow[k];
        qa[f][2 * j + 1] = (_Float16)qrow[k + 1];
      }
  }

  v8f oacc[4];
#pragma unroll
  for (int d = 0; d < 4; ++d) { v8f z = {}; oacc[d] = z; }
  float m_i[8], l_i[8];
#pragma unroll
  for (int i = 0; i < 8; ++i) { m_i[i] = -1.0e30f; l_i[i] = 0.0f; }

  stage(0, 0);  // prime the pipeline

  for (int kt = 0; kt < NKT; ++kt) {
    const int buf = kt & 1;
    // own stage(kt) copies done; barrier -> everyone's copies done AND
    // everyone finished compute(kt-1), so buf^1 is free to restage.
    asm volatile("s_wait_asynccnt 0x0" ::: "memory");
    __syncthreads();
    if (kt + 1 < NKT) stage(kt + 1, buf ^ 1);   // overlaps with compute below

    // ---- scores: S = Q K^T, four 16x16 sub-tiles over 64 keys ----
    v8f sc[4];
#pragma unroll
    for (int n = 0; n < 4; ++n) {
      const _Float16* kp = &Ks[buf][n * 16 + l16][hgrp * 16];
      const v16h kb0 = *(const v16h*)kp;
      const v16h kb1 = *(const v16h*)(kp + 32);
      v8f c = {};
      c = wmma_f16(qa[0], kb0, c);
      c = wmma_f16(qa[1], kb1, c);
      sc[n] = c;
    }

    // ---- mask (pre-scale, as reference), scale, online softmax ----
    float alpha[8];
#pragma unroll
    for (int i = 0; i < 8; ++i) {
      const int q = wave * 16 + 8 * hgrp + i;
#pragma unroll
      for (int n = 0; n < 4; ++n) {
        const int k = n * 16 + l16;
        const unsigned int wrd = Ms[buf][(q << 4) + (k >> 2)];
        const bool keep = (wrd >> ((k & 3) * 8)) & 1u;
        const float s = keep ? (float)sc[n][i] : -10000.0f;
        sc[n][i] = s * 0.125f;
      }
      float r = fmaxf(fmaxf((float)sc[0][i], (float)sc[1][i]),
                      fmaxf((float)sc[2][i], (float)sc[3][i]));
#pragma unroll
      for (int off = 1; off < 16; off <<= 1)
        r = fmaxf(r, __shfl_xor(r, off, 32));
      const float mnew = fmaxf(m_i[i], r);
      float sum = 0.0f;
#pragma unroll
      for (int n = 0; n < 4; ++n) {
        const float p = __expf((float)sc[n][i] - mnew);
        sc[n][i] = p;
        sum += p;
      }
#pragma unroll
      for (int off = 1; off < 16; off <<= 1)
        sum += __shfl_xor(sum, off, 32);
      const float a = __expf(m_i[i] - mnew);
      l_i[i] = l_i[i] * a + sum;
      m_i[i] = mnew;
      alpha[i] = a;
    }
#pragma unroll
    for (int d = 0; d < 4; ++d)
#pragma unroll
      for (int i = 0; i < 8; ++i)
        oacc[d][i] = (float)oacc[d][i] * alpha[i];

    // ---- stage P through wave-private LDS: C layout -> A layout ----
#pragma unroll
    for (int n = 0; n < 4; ++n)
#pragma unroll
      for (int i = 0; i < 8; ++i)
        Ps[wave][8 * hgrp + i][n * 16 + l16] = (_Float16)(float)sc[n][i];
    asm volatile("s_wait_dscnt 0x0" ::: "memory");  // DS in-order per wave

    v16h pa[2];
    {
      const _Float16* prow = &Ps[wave][l16][0];
#pragma unroll
      for (int f = 0; f < 2; ++f)
#pragma unroll
        for (int j = 0; j < 8; ++j) {
          const int k = f * 32 + hgrp * 16 + j * 2;
          pa[f][2 * j]     = prow[k];
          pa[f][2 * j + 1] = prow[k + 1];
        }
    }

    // ---- O += P V ----
#pragma unroll
    for (int d = 0; d < 4; ++d) {
      const _Float16* vp = &Vs[buf][d * 16 + l16][hgrp * 16];
      const v16h vb0 = *(const v16h*)vp;
      const v16h vb1 = *(const v16h*)(vp + 32);
      v8f c = oacc[d];
      c = wmma_f16(pa[0], vb0, c);
      c = wmma_f16(pa[1], vb1, c);
      oacc[d] = c;
    }
  }

  // ---- epilogue: normalize by row sum, store fp32 ----
#pragma unroll
  for (int i = 0; i < 8; ++i) {
    const float inv = 1.0f / l_i[i];
    float* orow = Ob + (size_t)(qrow0 + 8 * hgrp + i) * kD;
#pragma unroll
    for (int d = 0; d < 4; ++d)
      orow[d * 16 + l16] = (float)oacc[d][i] * inv;
  }
}

// ---------------------------------------------------------------------------
// Fallback (self-contained, no workspace): round-1 kernel, kept for the case
// ws_size < 33.5 MB. Same math, synchronous f32 staging with in-loop convert.
// ---------------------------------------------------------------------------
__launch_bounds__(NTHREADS, 1)
__global__ void fa_fwd_sync(const float* __restrict__ Qg,
                            const float* __restrict__ Kg,
                            const float* __restrict__ Vg,
                            const unsigned char* __restrict__ Mg,
                            float* __restrict__ Og)
{
  __shared__ _Float16     Ks[BN][kD];
  __shared__ _Float16     Vs[kD][BN];
  __shared__ unsigned int Ms[BM * BN / 4];
  __shared__ _Float16     Ps[WAVES][16][BN];

  const int tid  = threadIdx.x;
  const int wave = tid >> 5;
  const int lane = tid & 31;
  const int hgrp = lane >> 4;
  const int l16  = lane & 15;

  const int bh    = blockIdx.y;
  const int b     = bh / kH;
  const int qblk  = blockIdx.x * BM;
  const int qrow0 = qblk + wave * 16;

  const float* Qb = Qg + (size_t)bh * kS * kD;
  const float* Kb = Kg + (size_t)bh * kS * kD;
  const float* Vb = Vg + (size_t)bh * kS * kD;
  const unsigned char* Mb = Mg + (size_t)b * kS * kS;
  float* Ob = Og + (size_t)bh * kS * kD;

  v16h qa[2];
  {
    const float* qrow = Qb + (size_t)(qrow0 + l16) * kD;
#pragma unroll
    for (int f = 0; f < 2; ++f)
#pragma unroll
      for (int j = 0; j < 8; ++j) {
        const int k = f * 32 + hgrp * 16 + j * 2;
        qa[f][2 * j]     = (_Float16)qrow[k];
        qa[f][2 * j + 1] = (_Float16)qrow[k + 1];
      }
  }

  v8f oacc[4];
#pragma unroll
  for (int d = 0; d < 4; ++d) { v8f z = {}; oacc[d] = z; }
  float m_i[8], l_i[8];
#pragma unroll
  for (int i = 0; i < 8; ++i) { m_i[i] = -1.0e30f; l_i[i] = 0.0f; }

  for (int kt = 0; kt < NKT; ++kt) {
    const int kbase = kt * BN;
    for (int it = tid; it < BN * kD; it += NTHREADS) {
      const int key = it >> 6, d = it & 63;
      Ks[key][d] = (_Float16)Kb[(size_t)(kbase + key) * kD + d];
      Vs[d][key] = (_Float16)Vb[(size_t)(kbase + key) * kD + d];
    }
    for (int it = tid; it < BM * BN / 4; it += NTHREADS) {
      const int q = it >> 4, w = it & 15;
      Ms[it] = *(const unsigned int*)(Mb + (size_t)(qblk + q) * kS + kbase + w * 4);
    }
    __syncthreads();
    if (kt + 1 < NKT) {
      __builtin_prefetch(Kb + (size_t)(kbase + BN) * kD + tid * 16, 0, 1);
      __builtin_prefetch(Vb + (size_t)(kbase + BN) * kD + tid * 16, 0, 1);
    }

    v8f sc[4];
#pragma unroll
    for (int n = 0; n < 4; ++n) {
      const _Float16* kp = &Ks[n * 16 + l16][hgrp * 16];
      const v16h kb0 = *(const v16h*)kp;
      const v16h kb1 = *(const v16h*)(kp + 32);
      v8f c = {};
      c = wmma_f16(qa[0], kb0, c);
      c = wmma_f16(qa[1], kb1, c);
      sc[n] = c;
    }

    float alpha[8];
#pragma unroll
    for (int i = 0; i < 8; ++i) {
      const int q = wave * 16 + 8 * hgrp + i;
#pragma unroll
      for (int n = 0; n < 4; ++n) {
        const int k = n * 16 + l16;
        const unsigned int wrd = Ms[(q << 4) + (k >> 2)];
        const bool keep = (wrd >> ((k & 3) * 8)) & 1u;
        const float s = keep ? (float)sc[n][i] : -10000.0f;
        sc[n][i] = s * 0.125f;
      }
      float r = fmaxf(fmaxf((float)sc[0][i], (float)sc[1][i]),
                      fmaxf((float)sc[2][i], (float)sc[3][i]));
#pragma unroll
      for (int off = 1; off < 16; off <<= 1)
        r = fmaxf(r, __shfl_xor(r, off, 32));
      const float mnew = fmaxf(m_i[i], r);
      float sum = 0.0f;
#pragma unroll
      for (int n = 0; n < 4; ++n) {
        const float p = __expf((float)sc[n][i] - mnew);
        sc[n][i] = p;
        sum += p;
      }
#pragma unroll
      for (int off = 1; off < 16; off <<= 1)
        sum += __shfl_xor(sum, off, 32);
      const float a = __expf(m_i[i] - mnew);
      l_i[i] = l_i[i] * a + sum;
      m_i[i] = mnew;
      alpha[i] = a;
    }
#pragma unroll
    for (int d = 0; d < 4; ++d)
#pragma unroll
      for (int i = 0; i < 8; ++i)
        oacc[d][i] = (float)oacc[d][i] * alpha[i];

#pragma unroll
    for (int n = 0; n < 4; ++n)
#pragma unroll
      for (int i = 0; i < 8; ++i)
        Ps[wave][8 * hgrp + i][n * 16 + l16] = (_Float16)(float)sc[n][i];
    asm volatile("s_wait_dscnt 0x0" ::: "memory");

    v16h pa[2];
    {
      const _Float16* prow = &Ps[wave][l16][0];
#pragma unroll
      for (int f = 0; f < 2; ++f)
#pragma unroll
        for (int j = 0; j < 8; ++j) {
          const int k = f * 32 + hgrp * 16 + j * 2;
          pa[f][2 * j]     = prow[k];
          pa[f][2 * j + 1] = prow[k + 1];
        }
    }

#pragma unroll
    for (int d = 0; d < 4; ++d) {
      const _Float16* vp = &Vs[d * 16 + l16][hgrp * 16];
      const v16h vb0 = *(const v16h*)vp;
      const v16h vb1 = *(const v16h*)(vp + 32);
      v8f c = oacc[d];
      c = wmma_f16(pa[0], vb0, c);
      c = wmma_f16(pa[1], vb1, c);
      oacc[d] = c;
    }
    __syncthreads();
  }

#pragma unroll
  for (int i = 0; i < 8; ++i) {
    const float inv = 1.0f / l_i[i];
    float* orow = Ob + (size_t)(qrow0 + 8 * hgrp + i) * kD;
#pragma unroll
    for (int d = 0; d < 4; ++d)
      orow[d * 16 + l16] = (float)oacc[d][i] * inv;
  }
}

extern "C" void kernel_launch(void* const* d_in, const int* in_sizes, int n_in,
                              void* d_out, int out_size, void* d_ws, size_t ws_size,
                              hipStream_t stream) {
  (void)in_sizes; (void)n_in; (void)out_size;
  const float*         Q = (const float*)d_in[0];
  const float*         K = (const float*)d_in[1];
  const float*         V = (const float*)d_in[2];
  const unsigned char* M = (const unsigned char*)d_in[3];  // jnp.bool_ = 1 byte/elem
  float*               O = (float*)d_out;

  dim3 grid(kS / BM, kBH);
  dim3 block(NTHREADS);

  const size_t halves = (size_t)kBH * kS * kD;   // per tensor
  const size_t need   = 2 * halves * 2;          // K f16 + V f16 bytes (33.5 MB)

  if (ws_size >= need && d_ws != nullptr) {
    _Float16* Kh = (_Float16*)d_ws;
    _Float16* Vt = Kh + halves;
    cvt_kv<<<dim3(kS / 64, kBH), dim3(256), 0, stream>>>(K, V, Kh, Vt);
    fa_fwd_async<<<grid, block, 0, stream>>>(Q, Kh, Vt, M, O);
  } else {
    fa_fwd_sync<<<grid, block, 0, stream>>>(Q, K, V, M, O);
  }
}
